// HybridModel_85572928405789
// MI455X (gfx1250) — compile-verified, matrix operand order checked by reading, not run on hardware
//
#include <hip/hip_runtime.h>
#include <hip/hip_bf16.h>

typedef __attribute__((ext_vector_type(16))) _Float16 v16h;
typedef __attribute__((ext_vector_type(8)))  _Float16 v8h;
typedef __attribute__((ext_vector_type(8)))  float    v8f;

#define BATCH 2048

// Input-source modes for the conv staging loop
#define SRC_F16 0   // read f16 activations
#define SRC_F32 1   // read f32 tensor, convert (network input)
#define SRC_FC2 2   // compute rank-4 fc2 output on the fly (fused fc2 -> dconv1)

// ---------------------------------------------------------------------------
// K-ordering: k = p*CIN + ci  with p = ky*3+kx (spatial tap), ci innermost.
// This makes a lane's B-fragment a CONTIGUOUS run of channels at one spatial
// position when LDS is stored channel-interleaved, so the im2col gather
// becomes 1-2 vector ds loads instead of 16 scalar ones.  The A table below
// applies the identical permutation, so the GEMM result is unchanged.
// ---------------------------------------------------------------------------

// Prep: zero-padded f16 weights in WMMA A-fragment lane order:
//   dst[(chunk*32 + lane)*16 + i], row = lane&15, k = chunk*32+(lane>>4)*16+i
//   source value = w[row][ci][p] with ci = k % CIN, p = k / CIN
template<int COUT, int CIN>
__global__ void prep_afrag(const float* __restrict__ w, _Float16* __restrict__ dst) {
    constexpr int K      = CIN * 9;
    constexpr int CHUNKS = (K + 31) / 32;
    const int t = blockIdx.x * blockDim.x + threadIdx.x;
    if (t >= CHUNKS * 512) return;
    const int i    = t & 15;
    const int lane = (t >> 4) & 31;
    const int c    = t >> 9;
    const int row  = lane & 15;
    const int k    = c * 32 + (lane >> 4) * 16 + i;
    _Float16 v = (_Float16)0.0f;
    if (row < COUT && k < K) {
        const int ci = k % CIN;
        const int p  = k / CIN;
        v = (_Float16)w[(row * CIN + ci) * 9 + p];
    }
    dst[t] = v;
}

// ---------------------------------------------------------------------------
// 3x3 SAME conv as implicit GEMM on WMMA f16->f32.
// Block = 256 threads = 8 waves, computes 4 output rows of one image:
//   blockIdx.x = b*8 + ygroup ; rows y0..y0+3
//   wave wv: row = y0 + (wv>>1), pixels x in [16*(wv&1), 16*(wv&1)+16)
// Stages a 6-row halo band channel-interleaved: smem[(ry*34+xx)*CIN + ci].
// ACT: 0 = bias+ReLU f16 out ; 1 = bias+sigmoid f32 out (final layer).
// ---------------------------------------------------------------------------
template<int CIN, int COUT, int ACT, int SRC>
__global__ __launch_bounds__(256) void conv3x3_wmma(
    const void*     __restrict__ in,    // f16 or f32 [B][CIN][32][32] (unused for SRC_FC2)
    const _Float16* __restrict__ aw,    // prepped A fragments [CHUNKS][32][16]
    const float*    __restrict__ bias,  // [COUT]
    const float*    __restrict__ qv,    // [B][4]      (SRC_FC2 only)
    const float*    __restrict__ fc2w,  // [8192][4]   (SRC_FC2 only)
    const float*    __restrict__ fc2b,  // [8192]      (SRC_FC2 only)
    void*           __restrict__ outv)  // f16 or f32 [B][COUT][32][32]
{
    constexpr int K      = CIN * 9;
    constexpr int CHUNKS = (K + 31) / 32;
    constexpr int NS     = CIN * 6 * 34;

    const int b  = blockIdx.x >> 3;
    const int y0 = (blockIdx.x & 7) * 4;
    const int tid = threadIdx.x;

    __shared__ alignas(32) _Float16 smem[NS];   // [6][34][CIN] channel-interleaved

    float q0 = 0.f, q1 = 0.f, q2 = 0.f, q3 = 0.f;
    if (SRC == SRC_FC2) {
        q0 = qv[b * 4 + 0]; q1 = qv[b * 4 + 1];
        q2 = qv[b * 4 + 2]; q3 = qv[b * 4 + 3];
    }

    // Stage: iterate (ci, ry, xx) so global reads stay coalesced in xx,
    // write channel-interleaved into LDS.
    for (int t = tid; t < NS; t += 256) {
        const int xx  = t % 34;
        const int rem = t / 34;
        const int ry  = rem % 6;
        const int ci  = rem / 6;
        const int yy  = y0 + ry - 1;
        const int x   = xx - 1;
        _Float16 v = (_Float16)0.0f;
        if (yy >= 0 && yy < 32 && x >= 0 && x < 32) {
            const int idx = ((b * CIN + ci) * 32 + yy) * 32 + x;
            if (SRC == SRC_F16) {
                v = ((const _Float16*)in)[idx];
            } else if (SRC == SRC_F32) {
                v = (_Float16)(((const float*)in)[idx]);
            } else {            // fused fc2: g[b][j] = fc2_b[j] + sum_k q[k]*fc2_w[j][k]
                const int j = (ci * 32 + yy) * 32 + x;
                const float* wj = fc2w + j * 4;
                float f = fc2b[j] + q0 * wj[0] + q1 * wj[1] + q2 * wj[2] + q3 * wj[3];
                v = (_Float16)f;
            }
        }
        smem[(ry * 34 + xx) * CIN + ci] = v;
    }
    __syncthreads();

    const int lane  = tid & 31;
    const int wv    = tid >> 5;
    const int ryw   = wv >> 1;             // row within 4-row group
    const int y     = y0 + ryw;
    const int x0    = (wv & 1) * 16;
    const int col   = lane & 15;           // A row (out ch) / B col (pixel)
    const int h     = lane >> 4;           // K-half selector
    const int khalf = h * 16;

    v8f acc = {};
    #pragma unroll
    for (int c = 0; c < CHUNKS; ++c) {
        // A fragment: one contiguous 32-byte vector load
        const v16h afrag = *(const v16h*)(aw + (c * 32 + lane) * 16);

        // B fragment: contiguous channel runs at spatial tap(s) p
        v16h bfrag = {};
        if constexpr (CIN == 16) {
            // 16 elements = all channels at one tap: p = 2c + h
            const int p = 2 * c + h;
            if (p < 9) {
                const int ky = p / 3, kx = p - 3 * ky;
                bfrag = *(const v16h*)&smem[((ryw + ky) * 34 + (x0 + col + kx)) * 16];
            }
        } else if constexpr (CIN == 8) {
            // 16 elements = channels at taps q and q+1: q = 4c + 2h
            const int q = 4 * c + 2 * h;
            v8h lo = {}, hi = {};
            if (q < 9) {
                const int ky = q / 3, kx = q - 3 * ky;
                lo = *(const v8h*)&smem[((ryw + ky) * 34 + (x0 + col + kx)) * 8];
            }
            if (q + 1 < 9) {
                const int ky = (q + 1) / 3, kx = (q + 1) - 3 * ky;
                hi = *(const v8h*)&smem[((ryw + ky) * 34 + (x0 + col + kx)) * 8];
            }
            #pragma unroll
            for (int i = 0; i < 8; ++i) { bfrag[i] = lo[i]; bfrag[8 + i] = hi[i]; }
        } else {
            // generic scalar gather (conv1, CIN=3, single chunk)
            #pragma unroll
            for (int i = 0; i < 16; ++i) {
                const int k = c * 32 + khalf + i;
                _Float16 bv = (_Float16)0.0f;
                if (k < K) {
                    const int ci = k % CIN;
                    const int p  = k / CIN;
                    const int ky = p / 3, kx = p - 3 * ky;
                    bv = smem[((ryw + ky) * 34 + (x0 + col + kx)) * CIN + ci];
                }
                bfrag[i] = bv;
            }
        }

        acc = __builtin_amdgcn_wmma_f32_16x16x32_f16(
            false, afrag, false, bfrag, (short)0, acc, false, false);
    }

    const int x = x0 + col;
    #pragma unroll
    for (int r = 0; r < 8; ++r) {
        const int co = r + 8 * h;              // C/D layout: M = r + 8*(lane>>4)
        if (co < COUT) {
            float v = acc[r] + bias[co];
            if (ACT == 0) {
                v = v > 0.0f ? v : 0.0f;
                ((_Float16*)outv)[((b * COUT + co) * 32 + y) * 32 + x] = (_Float16)v;
            } else {
                v = 1.0f / (1.0f + __expf(-v));
                ((float*)outv)[((b * COUT + co) * 32 + y) * 32 + x] = v;
            }
        }
    }
}

// ---------------------------------------------------------------------------
// FC1: ang[b][k] = sum_j h[b][j] * fc_w[k][j] + fc_b[k]   (8192 -> 4)
// ---------------------------------------------------------------------------
__global__ __launch_bounds__(256) void fc1_kernel(
    const _Float16* __restrict__ h,    // [B][8192] f16
    const float*    __restrict__ w,    // [4][8192]
    const float*    __restrict__ bias, // [4]
    float*          __restrict__ ang)  // [B][4]
{
    __shared__ float red[4][256];
    const int b = blockIdx.x, tid = threadIdx.x;
    const _Float16* hb = h + (size_t)b * 8192;
    float p0 = 0.f, p1 = 0.f, p2 = 0.f, p3 = 0.f;
    for (int j = tid; j < 8192; j += 256) {
        float hv = (float)hb[j];
        p0 += hv * w[j];
        p1 += hv * w[8192 + j];
        p2 += hv * w[2 * 8192 + j];
        p3 += hv * w[3 * 8192 + j];
    }
    red[0][tid] = p0; red[1][tid] = p1; red[2][tid] = p2; red[3][tid] = p3;
    __syncthreads();
    for (int s = 128; s > 0; s >>= 1) {
        if (tid < s) {
            #pragma unroll
            for (int k = 0; k < 4; ++k) red[k][tid] += red[k][tid + s];
        }
        __syncthreads();
    }
    if (tid < 4) ang[b * 4 + tid] = red[tid][0] + bias[tid];
}

// ---------------------------------------------------------------------------
// 4-qubit statevector sim: one thread per sample, 16 complex amps in VGPRs.
// ---------------------------------------------------------------------------
__device__ __forceinline__ float2 cmul(float2 a, float2 b) {
    return make_float2(a.x * b.x - a.y * b.y, a.x * b.y + a.y * b.x);
}
__device__ __forceinline__ float2 cadd(float2 a, float2 b) {
    return make_float2(a.x + b.x, a.y + b.y);
}

__global__ void quantum_kernel(const float* __restrict__ ang, // [B][4]
                               const float* __restrict__ W,   // [3][4][3]
                               float*       __restrict__ qout)// [B][4]
{
    const int b = blockIdx.x * blockDim.x + threadIdx.x;
    if (b >= BATCH) return;

    float2 st[16];
    #pragma unroll
    for (int i = 0; i < 16; ++i) st[i] = make_float2(0.f, 0.f);
    st[0].x = 1.f;

    // AngleEmbedding: RX(theta) = [[c, -i s],[-i s, c]]
    #pragma unroll
    for (int wq = 0; wq < 4; ++wq) {
        float th = ang[b * 4 + wq] * 0.5f;
        float c = __cosf(th), s = __sinf(th);
        const int bit = 1 << (3 - wq);
        #pragma unroll
        for (int i = 0; i < 16; ++i) {
            if (i & bit) continue;
            const int j = i | bit;
            float2 a0 = st[i], a1 = st[j];
            st[i] = make_float2(c * a0.x + s * a1.y, c * a0.y - s * a1.x);
            st[j] = make_float2(s * a0.y + c * a1.x, -s * a0.x + c * a1.y);
        }
    }

    // StronglyEntanglingLayers: Rot(phi,theta,omega) per wire + CNOT ring
    #pragma unroll
    for (int l = 0; l < 3; ++l) {
        #pragma unroll
        for (int wq = 0; wq < 4; ++wq) {
            float phi = W[(l * 4 + wq) * 3 + 0];
            float th  = W[(l * 4 + wq) * 3 + 1];
            float om  = W[(l * 4 + wq) * 3 + 2];
            float ct = __cosf(th * 0.5f), stt = __sinf(th * 0.5f);
            float a = (phi + om) * 0.5f, d = (phi - om) * 0.5f;
            float ca = __cosf(a), sa = __sinf(a);
            float cd = __cosf(d), sd = __sinf(d);
            float2 u00 = make_float2( ct * ca, -ct * sa);
            float2 u01 = make_float2(-stt * cd, -stt * sd);
            float2 u10 = make_float2( stt * cd, -stt * sd);
            float2 u11 = make_float2( ct * ca,  ct * sa);
            const int bit = 1 << (3 - wq);
            #pragma unroll
            for (int i = 0; i < 16; ++i) {
                if (i & bit) continue;
                const int j = i | bit;
                float2 a0 = st[i], a1 = st[j];
                st[i] = cadd(cmul(u00, a0), cmul(u01, a1));
                st[j] = cadd(cmul(u10, a0), cmul(u11, a1));
            }
        }
        const int r = (l % 3) + 1;
        #pragma unroll
        for (int wq = 0; wq < 4; ++wq) {
            const int t  = (wq + r) & 3;
            const int bc = 1 << (3 - wq);
            const int bt = 1 << (3 - t);
            #pragma unroll
            for (int i = 0; i < 16; ++i) {
                if ((i & bc) && !(i & bt)) {
                    const int j = i | bt;
                    float2 tmp = st[i]; st[i] = st[j]; st[j] = tmp;
                }
            }
        }
    }

    // PauliZ expectation values
    #pragma unroll
    for (int wq = 0; wq < 4; ++wq) {
        const int bit = 1 << (3 - wq);
        float ev = 0.f;
        #pragma unroll
        for (int i = 0; i < 16; ++i) {
            float p = st[i].x * st[i].x + st[i].y * st[i].y;
            ev += (i & bit) ? -p : p;
        }
        qout[b * 4 + wq] = ev;
    }
}

// ---------------------------------------------------------------------------
// Launcher
// ---------------------------------------------------------------------------
static inline size_t align256(size_t x) { return (x + 255) & ~(size_t)255; }

extern "C" void kernel_launch(void* const* d_in, const int* in_sizes, int n_in,
                              void* d_out, int out_size, void* d_ws, size_t ws_size,
                              hipStream_t stream) {
    const float* x        = (const float*)d_in[0];
    const float* conv1_w  = (const float*)d_in[1];
    const float* conv1_b  = (const float*)d_in[2];
    const float* conv2_w  = (const float*)d_in[3];
    const float* conv2_b  = (const float*)d_in[4];
    const float* fc_w     = (const float*)d_in[5];
    const float* fc_b     = (const float*)d_in[6];
    const float* q_w      = (const float*)d_in[7];
    const float* fc2_w    = (const float*)d_in[8];
    const float* fc2_b    = (const float*)d_in[9];
    const float* dconv1_w = (const float*)d_in[10];
    const float* dconv1_b = (const float*)d_in[11];
    const float* dconv2_w = (const float*)d_in[12];
    const float* dconv2_b = (const float*)d_in[13];
    float* out = (float*)d_out;
    (void)ws_size; (void)n_in; (void)in_sizes; (void)out_size;

    char* ws = (char*)d_ws;
    size_t o = 0;
    _Float16* bufA = (_Float16*)(ws + o); o += align256((size_t)BATCH * 16 * 1024 * 2); // h1 / h3
    _Float16* bufB = (_Float16*)(ws + o); o += align256((size_t)BATCH * 8  * 1024 * 2); // h2
    float*    ang  = (float*)(ws + o);    o += align256((size_t)BATCH * 4 * 4);
    float*    qv   = (float*)(ws + o);    o += align256((size_t)BATCH * 4 * 4);
    _Float16* a1   = (_Float16*)(ws + o); o += align256((size_t)1 * 512 * 2);  // conv1  K=27
    _Float16* a2   = (_Float16*)(ws + o); o += align256((size_t)5 * 512 * 2);  // conv2  K=144
    _Float16* a3   = (_Float16*)(ws + o); o += align256((size_t)3 * 512 * 2);  // dconv1 K=72
    _Float16* a4   = (_Float16*)(ws + o); o += align256((size_t)5 * 512 * 2);  // dconv2 K=144

    // Weight -> WMMA A-fragment tables (permuted K-order, tiny)
    prep_afrag<16, 3 ><<<(1 * 512 + 255) / 256, 256, 0, stream>>>(conv1_w,  a1);
    prep_afrag<8,  16><<<(5 * 512 + 255) / 256, 256, 0, stream>>>(conv2_w,  a2);
    prep_afrag<16, 8 ><<<(3 * 512 + 255) / 256, 256, 0, stream>>>(dconv1_w, a3);
    prep_afrag<3,  16><<<(5 * 512 + 255) / 256, 256, 0, stream>>>(dconv2_w, a4);

    const int convBlocks = BATCH * 8;    // 4 output rows per 8-wave block

    conv3x3_wmma<3, 16, 0, SRC_F32><<<convBlocks, 256, 0, stream>>>(
        (const void*)x, a1, conv1_b, nullptr, nullptr, nullptr, (void*)bufA);
    conv3x3_wmma<16, 8, 0, SRC_F16><<<convBlocks, 256, 0, stream>>>(
        (const void*)bufA, a2, conv2_b, nullptr, nullptr, nullptr, (void*)bufB);

    fc1_kernel<<<BATCH, 256, 0, stream>>>(bufB, fc_w, fc_b, ang);
    quantum_kernel<<<BATCH / 256, 256, 0, stream>>>(ang, q_w, qv);

    conv3x3_wmma<8, 16, 0, SRC_FC2><<<convBlocks, 256, 0, stream>>>(
        nullptr, a3, dconv1_b, qv, fc2_w, fc2_b, (void*)bufA);
    conv3x3_wmma<16, 3, 1, SRC_F16><<<convBlocks, 256, 0, stream>>>(
        (const void*)bufA, a4, dconv2_b, nullptr, nullptr, nullptr, (void*)out);
}
